// EncoderOnlyBlock_77816217469428
// MI455X (gfx1250) — matrix operand order repaired and checked by reading, not compiled, not run on hardware
//
#include <hip/hip_runtime.h>

// ---------------------------------------------------------------------------
// Types / helpers
// ---------------------------------------------------------------------------
typedef __bf16 bf16;
typedef __attribute__((ext_vector_type(16))) __bf16 bf16x16;
typedef __attribute__((ext_vector_type(8)))  __bf16 bf16x8;
typedef __attribute__((ext_vector_type(8)))  float  f32x8;
typedef int v4i_gcc __attribute__((vector_size(16)));   // matches builtin param

#define AS1 __attribute__((address_space(1)))
#define AS3 __attribute__((address_space(3)))

#if defined(__has_builtin)
#if __has_builtin(__builtin_amdgcn_global_load_async_to_lds_b128)
#define HAVE_ASYNC_LDS 1
#endif
#endif
#ifndef HAVE_ASYNC_LDS
#define HAVE_ASYNC_LDS 0
#endif

struct __align__(16) Chunk16 { unsigned int u[4]; };   // 16-byte copy unit

union ABFrag { bf16x16 v; bf16x8 h[2]; };

__device__ __forceinline__ bf16 f2bf(float f) {
    unsigned u = __builtin_bit_cast(unsigned, f);
    u += 0x7FFFu + ((u >> 16) & 1u);                   // round-to-nearest-even
    return __builtin_bit_cast(bf16, (unsigned short)(u >> 16));
}

__device__ __forceinline__ void wait_async0() {
#if HAVE_ASYNC_LDS
#if defined(__has_builtin) && __has_builtin(__builtin_amdgcn_s_wait_asynccnt)
    __builtin_amdgcn_s_wait_asynccnt(0);
#else
    asm volatile("s_wait_asynccnt 0x0" ::: "memory");
#endif
#endif
}

// ---------------------------------------------------------------------------
// fp32 -> bf16 flat convert (4 elems / thread)
// ---------------------------------------------------------------------------
__global__ __launch_bounds__(256) void cvt_f32_bf16(const float* __restrict__ in,
                                                    bf16* __restrict__ out) {
    size_t i = ((size_t)blockIdx.x * 256 + threadIdx.x) * 4;
#pragma unroll
    for (int j = 0; j < 4; ++j) out[i + j] = f2bf(in[i + j]);
}

// ---------------------------------------------------------------------------
// fp32 (R x C) -> bf16 (C x R) transpose-convert; blockIdx.z batches
// ---------------------------------------------------------------------------
__global__ __launch_bounds__(256) void transpose_cvt(const float* __restrict__ in,
                                                     bf16* __restrict__ out,
                                                     int R, int C) {
    size_t boff = (size_t)blockIdx.z * R * C;
    unsigned idx = blockIdx.x * 256u + threadIdx.x;
    unsigned r = idx / (unsigned)C;
    unsigned c = idx % (unsigned)C;
    out[boff + (size_t)c * R + r] = f2bf(in[boff + idx]);
}

// ---------------------------------------------------------------------------
// bf16 WMMA GEMM:  C[M,N] = scale * A[M,K] * Bt[N,K]^T (+ bias[N] if HB)
//   MODE 0: C -> f32          (Cf, ldc)
//   MODE 1: C -> bf16         (Cb, ldc)
//   MODE 2: C -> bf16, stored transposed: Cb[n*ldc + m]
// Block tile 128x128x32, 8 waves, wave tile 32x64 (2x4 WMMA frags).
// Double-buffered LDS stages filled by async global->LDS DMA (ASYNCcnt),
// falling back to ld+ds_store when the builtin is unavailable.
// M % 128 == 0, N % 128 == 0, K % 32 == 0 in every use here.
// ---------------------------------------------------------------------------
template <int MODE, bool HB>
__global__ __launch_bounds__(256) void gemm_nt(
    const bf16* __restrict__ A, int lda,
    const bf16* __restrict__ Bt, int ldb,
    float* __restrict__ Cf, bf16* __restrict__ Cb, int ldc,
    const float* __restrict__ bias,
    float scale, int M, int N, int K)
{
    constexpr int LDSD = 40;                 // 32 data cols + 8 pad (bank-safe)
    __shared__ bf16 As[2][128 * LDSD];       // 2 x 10 KB
    __shared__ bf16 Bs[2][128 * LDSD];       // 2 x 10 KB

    const int tid  = threadIdx.x;
    const int wid  = tid >> 5;
    const int lane = tid & 31;
    const int wm   = wid & 3;                // 4 waves along M
    const int wn   = wid >> 2;               // 2 waves along N
    const int m0   = blockIdx.y * 128;
    const int n0   = blockIdx.x * 128;

    const f32x8 zero = {0.f, 0.f, 0.f, 0.f, 0.f, 0.f, 0.f, 0.f};
    f32x8 acc[2][4];
#pragma unroll
    for (int mi = 0; mi < 2; ++mi)
#pragma unroll
        for (int ni = 0; ni < 4; ++ni) acc[mi][ni] = zero;

    const int laneRow = lane & 15;
    const int laneHi  = lane >> 4;

    // ---- stage loader: 128x32 bf16 tiles, 16B chunks, 2 chunks/thread ----
    auto issue = [&](int k0, int st) {
#pragma unroll
        for (int i = 0; i < 2; ++i) {
            int c   = tid + i * 256;
            int row = c >> 2;
            int col = (c & 3) << 3;
            const bf16* ga = A  + (size_t)(m0 + row) * lda + k0 + col;
            const bf16* gb = Bt + (size_t)(n0 + row) * ldb + k0 + col;
            bf16* la = &As[st][row * LDSD + col];
            bf16* lb = &Bs[st][row * LDSD + col];
#if HAVE_ASYNC_LDS
            __builtin_amdgcn_global_load_async_to_lds_b128(
                (AS1 v4i_gcc*)(v4i_gcc*)(void*)ga,
                (AS3 v4i_gcc*)(v4i_gcc*)(void*)la, 0, 0);
            __builtin_amdgcn_global_load_async_to_lds_b128(
                (AS1 v4i_gcc*)(v4i_gcc*)(void*)gb,
                (AS3 v4i_gcc*)(v4i_gcc*)(void*)lb, 0, 0);
#else
            *(Chunk16*)la = *(const Chunk16*)ga;
            *(Chunk16*)lb = *(const Chunk16*)gb;
#endif
        }
    };

    const int T = K >> 5;                    // K / 32 stages
    issue(0, 0);

    for (int t = 0; t < T; ++t) {
        wait_async0();                       // own stage-t DMA complete
        __syncthreads();                     // all waves: tile t resident,
                                             // tile (t-1) compute finished
        if (t + 1 < T) issue((t + 1) << 5, (t + 1) & 1);  // overlaps compute

        const bf16* as = As[t & 1];
        const bf16* bs = Bs[t & 1];

        ABFrag a[2], b[4];
#pragma unroll
        for (int mi = 0; mi < 2; ++mi) {
            // ISA 16-bit A layout: lanes<16 K=0..7 & 16..23, lanes>=16 +8
            int r   = wm * 32 + mi * 16 + laneRow;
            int off = r * LDSD + laneHi * 8;
            a[mi].h[0] = *(const bf16x8*)(&as[off]);
            a[mi].h[1] = *(const bf16x8*)(&as[off + 16]);
        }
#pragma unroll
        for (int ni = 0; ni < 4; ++ni) {
            // B layout: lanes<16 K=0..7 & 16..23, lanes>=16 +8 (col N = lane)
            int r   = wn * 64 + ni * 16 + laneRow;
            int off = r * LDSD + laneHi * 8;
            b[ni].h[0] = *(const bf16x8*)(&bs[off]);
            b[ni].h[1] = *(const bf16x8*)(&bs[off + 16]);
        }
#pragma unroll
        for (int mi = 0; mi < 2; ++mi)
#pragma unroll
            for (int ni = 0; ni < 4; ++ni)
                acc[mi][ni] = __builtin_amdgcn_wmma_f32_16x16x32_bf16(
                    false, a[mi].v, false, b[ni].v,
                    (short)0, acc[mi][ni], false, false);
    }

    // ---- epilogue: C frag = VGPR i -> (M = i + 8*laneHi, N = lane&15) ----
    const int nT = lane & 15;
    const int mT = laneHi << 3;
#pragma unroll
    for (int mi = 0; mi < 2; ++mi)
#pragma unroll
        for (int ni = 0; ni < 4; ++ni) {
            int gm = m0 + wm * 32 + mi * 16 + mT;
            int gn = n0 + wn * 64 + ni * 16 + nT;
            float bb = HB ? bias[gn] : 0.f;
#pragma unroll
            for (int i = 0; i < 8; ++i) {
                float v = acc[mi][ni][i] * scale + bb;
                if constexpr (MODE == 0)
                    Cf[(size_t)(gm + i) * ldc + gn] = v;
                else if constexpr (MODE == 1)
                    Cb[(size_t)(gm + i) * ldc + gn] = f2bf(v);
                else
                    Cb[(size_t)gn * ldc + (gm + i)] = f2bf(v);
            }
        }
}

// ---------------------------------------------------------------------------
// Row softmax: P[r,:] = softmax(S[r,:]) -> bf16. One block per row.
// ---------------------------------------------------------------------------
__global__ __launch_bounds__(256) void softmax_rows(const float* __restrict__ S,
                                                    bf16* __restrict__ P, int n) {
    const int row = blockIdx.x;
    const float* s = S + (size_t)row * n;
    bf16* p = P + (size_t)row * n;
    __shared__ float red[256];

    float m = -3.0e38f;
    for (int c = threadIdx.x; c < n; c += 256) m = fmaxf(m, s[c]);
    red[threadIdx.x] = m; __syncthreads();
    for (int st = 128; st > 0; st >>= 1) {
        if (threadIdx.x < st) red[threadIdx.x] = fmaxf(red[threadIdx.x], red[threadIdx.x + st]);
        __syncthreads();
    }
    m = red[0]; __syncthreads();

    float sum = 0.f;
    for (int c = threadIdx.x; c < n; c += 256) sum += __expf(s[c] - m);
    red[threadIdx.x] = sum; __syncthreads();
    for (int st = 128; st > 0; st >>= 1) {
        if (threadIdx.x < st) red[threadIdx.x] += red[threadIdx.x + st];
        __syncthreads();
    }
    const float inv = 1.f / red[0];
    for (int c = threadIdx.x; c < n; c += 256) p[c] = f2bf(__expf(s[c] - m) * inv);
}

// ---------------------------------------------------------------------------
// y = LayerNorm(x + r) * g + be ; optional bf16 copy. One block per row, D=1024
// ---------------------------------------------------------------------------
__global__ __launch_bounds__(256) void add_ln(const float* __restrict__ xin,
                                              const float* __restrict__ r,
                                              const float* __restrict__ g,
                                              const float* __restrict__ be,
                                              float* __restrict__ yout,
                                              bf16* __restrict__ yh, int D) {
    const size_t base = (size_t)blockIdx.x * D;
    const int c0 = threadIdx.x * 4;
    float v[4], s = 0.f, ss = 0.f;
#pragma unroll
    for (int i = 0; i < 4; ++i) {
        float t = xin[base + c0 + i] + r[base + c0 + i];
        v[i] = t; s += t; ss += t * t;
    }
    __shared__ float rs[256], rss[256];
    rs[threadIdx.x] = s; rss[threadIdx.x] = ss; __syncthreads();
    for (int st = 128; st > 0; st >>= 1) {
        if (threadIdx.x < st) { rs[threadIdx.x] += rs[threadIdx.x + st];
                                rss[threadIdx.x] += rss[threadIdx.x + st]; }
        __syncthreads();
    }
    const float mu  = rs[0] / D;
    const float var = rss[0] / D - mu * mu;
    const float k   = rsqrtf(var + 1e-5f);
#pragma unroll
    for (int i = 0; i < 4; ++i) {
        float o = (v[i] - mu) * k * g[c0 + i] + be[c0 + i];
        yout[base + c0 + i] = o;
        if (yh) yh[base + c0 + i] = f2bf(o);
    }
}

// ---------------------------------------------------------------------------
// Host orchestration
// ---------------------------------------------------------------------------
extern "C" void kernel_launch(void* const* d_in, const int* in_sizes, int n_in,
                              void* d_out, int out_size, void* d_ws, size_t ws_size,
                              hipStream_t stream) {
    constexpr int D_ = 1024, H_ = 4, B_ = 4, S_ = 2048, M_ = B_ * S_; // 8192

    const float* x   = (const float*)d_in[0];
    const float* Wq  = (const float*)d_in[1];
    const float* bq  = (const float*)d_in[2];
    const float* Wk  = (const float*)d_in[3];
    const float* bk  = (const float*)d_in[4];
    const float* Wv  = (const float*)d_in[5];
    const float* bv  = (const float*)d_in[6];
    const float* W1  = (const float*)d_in[7];
    const float* b1  = (const float*)d_in[8];
    const float* g1  = (const float*)d_in[9];
    const float* be1 = (const float*)d_in[10];
    const float* W2  = (const float*)d_in[11];
    const float* b2  = (const float*)d_in[12];
    const float* g2  = (const float*)d_in[13];
    const float* be2 = (const float*)d_in[14];

    char* ws = (char*)d_ws;
    size_t off = 0;
    auto carve = [&](size_t bytes) -> void* {
        void* p = ws + off;
        off += (bytes + 255) & ~(size_t)255;
        return p;
    };

    bf16*  xh     = (bf16*) carve((size_t)M_ * D_ * 2);          // 16 MB
    bf16*  Wqt    = (bf16*) carve((size_t)H_ * D_ * D_ * 2);     //  8 MB
    bf16*  Wkt    = (bf16*) carve((size_t)H_ * D_ * D_ * 2);     //  8 MB
    bf16*  Wvt    = (bf16*) carve((size_t)H_ * D_ * D_ * 2);     //  8 MB
    bf16*  W1t    = (bf16*) carve((size_t)(H_ * D_) * D_ * 2);   //  8 MB
    bf16*  W2t    = (bf16*) carve((size_t)D_ * D_ * 2);          //  2 MB
    bf16*  Qh     = (bf16*) carve((size_t)M_ * D_ * 2);          // 16 MB (per-h)
    bf16*  Kh     = (bf16*) carve((size_t)M_ * D_ * 2);          // 16 MB (per-h)
    bf16*  Vt     = (bf16*) carve((size_t)M_ * D_ * 2);          // 16 MB (per-h, DxM)
    float* Sc     = (float*) carve((size_t)S_ * S_ * 4);         // 16 MB (per h,b)
    bf16*  P      = (bf16*) carve((size_t)S_ * S_ * 2);          //  8 MB (per h,b)
    bf16*  concat = (bf16*) carve((size_t)M_ * (H_ * D_) * 2);   // 64 MB
    float* proj   = (float*) carve((size_t)M_ * D_ * 4);         // 32 MB (reused as z)
    float* y      = (float*) carve((size_t)M_ * D_ * 4);         // 32 MB
    bf16*  yh     = xh;   // xh dead after projections -> reuse

    auto gemm = [&](int mode, const bf16* A, int lda, const bf16* Bt, int ldb,
                    void* C, int ldc, const float* bias, float scale,
                    int M, int N, int K) {
        dim3 g(N / 128, M / 128, 1), blk(256);
        if (mode == 0) {
            if (bias) gemm_nt<0, true ><<<g, blk, 0, stream>>>(A, lda, Bt, ldb,
                (float*)C, nullptr, ldc, bias, scale, M, N, K);
            else      gemm_nt<0, false><<<g, blk, 0, stream>>>(A, lda, Bt, ldb,
                (float*)C, nullptr, ldc, nullptr, scale, M, N, K);
        } else if (mode == 1) {
            if (bias) gemm_nt<1, true ><<<g, blk, 0, stream>>>(A, lda, Bt, ldb,
                nullptr, (bf16*)C, ldc, bias, scale, M, N, K);
            else      gemm_nt<1, false><<<g, blk, 0, stream>>>(A, lda, Bt, ldb,
                nullptr, (bf16*)C, ldc, nullptr, scale, M, N, K);
        } else {
            if (bias) gemm_nt<2, true ><<<g, blk, 0, stream>>>(A, lda, Bt, ldb,
                nullptr, (bf16*)C, ldc, bias, scale, M, N, K);
            else      gemm_nt<2, false><<<g, blk, 0, stream>>>(A, lda, Bt, ldb,
                nullptr, (bf16*)C, ldc, nullptr, scale, M, N, K);
        }
    };

    // --- 1. bf16 conversions ------------------------------------------------
    cvt_f32_bf16<<<dim3((M_ * D_) / 1024), dim3(256), 0, stream>>>(x, xh);
    transpose_cvt<<<dim3((D_ * D_) / 256, 1, H_), dim3(256), 0, stream>>>(Wq, Wqt, D_, D_);
    transpose_cvt<<<dim3((D_ * D_) / 256, 1, H_), dim3(256), 0, stream>>>(Wk, Wkt, D_, D_);
    transpose_cvt<<<dim3((D_ * D_) / 256, 1, H_), dim3(256), 0, stream>>>(Wv, Wvt, D_, D_);
    transpose_cvt<<<dim3((H_ * D_ * D_) / 256, 1, 1), dim3(256), 0, stream>>>(W1, W1t, H_ * D_, D_);
    transpose_cvt<<<dim3((D_ * D_) / 256, 1, 1), dim3(256), 0, stream>>>(W2, W2t, D_, D_);

    // --- 2. per-head attention ---------------------------------------------
    const float iscale = 0.03125f;  // 1/sqrt(1024)
    for (int h = 0; h < H_; ++h) {
        const bf16* wq = Wqt + (size_t)h * D_ * D_;
        const bf16* wk = Wkt + (size_t)h * D_ * D_;
        const bf16* wv = Wvt + (size_t)h * D_ * D_;
        gemm(1, xh, D_, wq, D_, Qh, D_,   bq + h * D_, 1.f, M_, D_, D_);
        gemm(1, xh, D_, wk, D_, Kh, D_,   bk + h * D_, 1.f, M_, D_, D_);
        gemm(2, xh, D_, wv, D_, Vt, M_,   bv + h * D_, 1.f, M_, D_, D_); // V^T (D x M)

        for (int b = 0; b < B_; ++b) {
            const bf16* Qb = Qh + (size_t)b * S_ * D_;
            const bf16* Kb = Kh + (size_t)b * S_ * D_;
            // scores = scale * Q @ K^T  (f32)
            gemm(0, Qb, D_, Kb, D_, Sc, S_, nullptr, iscale, S_, S_, D_);
            softmax_rows<<<dim3(S_), dim3(256), 0, stream>>>(Sc, P, S_);
            // head = P @ V  -> written directly into concat[b, :, h*D : (h+1)*D]
            bf16* Cp = concat + (size_t)b * S_ * (H_ * D_) + (size_t)h * D_;
            gemm(1, P, S_, Vt + (size_t)b * S_, M_, Cp, H_ * D_, nullptr, 1.f,
                 S_, D_, S_);
        }
    }

    // --- 3. output projection + LN1 -----------------------------------------
    gemm(0, concat, H_ * D_, W1t, H_ * D_, proj, D_, b1, 1.f, M_, D_, H_ * D_);
    add_ln<<<dim3(M_), dim3(256), 0, stream>>>(x, proj, g1, be1, y, yh, D_);

    // --- 4. FFN-ish W2 + LN2 -> d_out ---------------------------------------
    gemm(0, yh, D_, W2t, D_, proj /*z*/, D_, b2, 1.f, M_, D_, D_);
    add_ln<<<dim3(M_), dim3(256), 0, stream>>>(y, proj, g2, be2, (float*)d_out,
                                               nullptr, D_);
}
